// RegularNet_41772851921232
// MI455X (gfx1250) — matrix operand order repaired and checked by reading, not compiled, not run on hardware
//
#include <hip/hip_runtime.h>
#include <hip/hip_bf16.h>

#define CC       66
#define TT       96
#define BATCH    2048
#define FIN      7128      // 9*66*12
#define FIN_PAD  7136      // next multiple of 32
#define NH       1936
#define NH_PAD   1952      // next multiple of 32
#define NCLS     14

typedef _Float16     h16  __attribute__((ext_vector_type(16)));
typedef _Float16     h4   __attribute__((ext_vector_type(4)));
typedef float        f8   __attribute__((ext_vector_type(8)));
typedef float        f4   __attribute__((ext_vector_type(4)));
typedef unsigned int u4   __attribute__((ext_vector_type(4)));

// ---------------------------------------------------------------------------
// Kernel 1: fused per-(batch,channel) conv feature extractor.
// One thread handles one (b,c) pipeline; 32-thread (one wave32) blocks.
// LDS is lane-private columns: lds[elem*32 + lane]  (conflict-free).
//   R: 96 floats (input row)       offset 0
//   S: 384 floats (8 x 48 stage1)  offset 96*32
//   M: 96 floats (4 x 24 stage2)   offset 480*32
// Output features written directly as f16 (A-operand of the fc1 WMMA GEMM).
// ---------------------------------------------------------------------------
template<int K>
__device__ __forceinline__ void branch_eval(
    float* __restrict__ lds, int lane, int c,
    const float* __restrict__ ws,  const float* __restrict__ bs,
    const float* __restrict__ wi1, const float* __restrict__ bi1,
    const float* __restrict__ wi2, const float* __restrict__ bi2,
    _Float16* __restrict__ fo)
{
  constexpr int PAD = K / 2;
  float* R = lds;
  float* S = lds + 96 * 32;
  float* M = lds + 480 * 32;

  // stage 1: shared conv 1->8 (K taps, pad), relu, pool2 : R(96) -> S[8][48]
  for (int o = 0; o < 8; ++o) {
    float w[K];
#pragma unroll
    for (int k = 0; k < K; ++k) w[k] = ws[o * K + k];
    const float bo = bs[o];
    for (int q = 0; q < 48; ++q) {
      float a0 = bo, a1 = bo;
#pragma unroll
      for (int k = 0; k < K; ++k) {
        const int t0 = 2 * q + k - PAD;
        const int t1 = t0 + 1;
        if (t0 >= 0 && t0 < 96) a0 += w[k] * R[t0 * 32 + lane];
        if (t1 >= 0 && t1 < 96) a1 += w[k] * R[t1 * 32 + lane];
      }
      S[(o * 48 + q) * 32 + lane] = 0.5f * (fmaxf(a0, 0.f) + fmaxf(a1, 0.f));
    }
  }

  // stage 2: per-channel grouped conv 8->4, relu, pool2 : S -> M[4][24]
  for (int j = 0; j < 4; ++j) {
    float w[8 * K];
#pragma unroll
    for (int i = 0; i < 8 * K; ++i) w[i] = wi1[(c * 4 + j) * (8 * K) + i];
    const float bj = bi1[c * 4 + j];
    for (int p = 0; p < 24; ++p) {
      float a0 = bj, a1 = bj;
#pragma unroll
      for (int i = 0; i < 8; ++i) {
#pragma unroll
        for (int k = 0; k < K; ++k) {
          const int t0 = 2 * p + k - PAD;
          const int t1 = t0 + 1;
          if (t0 >= 0 && t0 < 48) a0 += w[i * K + k] * S[(i * 48 + t0) * 32 + lane];
          if (t1 >= 0 && t1 < 48) a1 += w[i * K + k] * S[(i * 48 + t1) * 32 + lane];
        }
      }
      M[(j * 24 + p) * 32 + lane] = 0.5f * (fmaxf(a0, 0.f) + fmaxf(a1, 0.f));
    }
  }

  // stage 3: per-channel grouped conv 4->4, relu, pool2 : M -> fo[4][12] (f16)
  for (int j = 0; j < 4; ++j) {
    float w[4 * K];
#pragma unroll
    for (int i = 0; i < 4 * K; ++i) w[i] = wi2[(c * 4 + j) * (4 * K) + i];
    const float bj = bi2[c * 4 + j];
    for (int p = 0; p < 12; ++p) {
      float a0 = bj, a1 = bj;
#pragma unroll
      for (int i = 0; i < 4; ++i) {
#pragma unroll
        for (int k = 0; k < K; ++k) {
          const int t0 = 2 * p + k - PAD;
          const int t1 = t0 + 1;
          if (t0 >= 0 && t0 < 24) a0 += w[i * K + k] * M[(i * 24 + t0) * 32 + lane];
          if (t1 >= 0 && t1 < 24) a1 += w[i * K + k] * M[(i * 24 + t1) * 32 + lane];
        }
      }
      fo[j * 12 + p] = (_Float16)(0.5f * (fmaxf(a0, 0.f) + fmaxf(a1, 0.f)));
    }
  }
}

__global__ __launch_bounds__(32) void feat_kernel(
    const float* __restrict__ x,
    const float* __restrict__ w_hs,  const float* __restrict__ b_hs,
    const float* __restrict__ w_ls,  const float* __restrict__ b_ls,
    const float* __restrict__ w_hi1, const float* __restrict__ b_hi1,
    const float* __restrict__ w_hi2, const float* __restrict__ b_hi2,
    const float* __restrict__ w_li1, const float* __restrict__ b_li1,
    const float* __restrict__ w_li2, const float* __restrict__ b_li2,
    _Float16* __restrict__ feat)   // [BATCH][FIN_PAD]
{
  extern __shared__ float lds[];            // 576 * 32 floats = 73728 B
  const int lane = threadIdx.x;
  const int bc = blockIdx.x * 32 + lane;    // grid covers BATCH*CC exactly
  const int b = bc / CC;
  const int c = bc - b * CC;

  // load this (b,c) temporal row (consecutive lanes -> consecutive c: coalesced)
  const float* xp = x + (size_t)b * TT * CC + c;
#pragma unroll 4
  for (int t = 0; t < TT; ++t) lds[t * 32 + lane] = xp[t * CC];

  _Float16* fo = feat + (size_t)b * FIN_PAD + c * 108;

  // "orig" row: pool2^3 = mean of 8 consecutive samples
  for (int t = 0; t < 12; ++t) {
    float s = 0.f;
#pragma unroll
    for (int u = 0; u < 8; ++u) s += lds[(t * 8 + u) * 32 + lane];
    fo[t] = (_Float16)(s * 0.125f);
  }
  // concat order per channel: [orig(1), low(4), high(4)] x 12
  branch_eval<3>(lds, lane, c, w_ls, b_ls, w_li1, b_li1, w_li2, b_li2, fo + 12);
  branch_eval<7>(lds, lane, c, w_hs, b_hs, w_hi1, b_hi1, w_hi2, b_hi2, fo + 60);
}

// ---------------------------------------------------------------------------
// Kernel 2: fc1 GEMM via WMMA f16 (f32 accumulate), fused bias+ReLU+f16 cvt.
//   out[m][n] = relu( sum_k feat[m][k] * w_fc1[n][k] + b1[n] )
// Block: 256 threads (8 waves), tile BM=128 x BN=64, BK=32.
// LDS A tile [m][k] f16 (8KB), B tile [n][k] f16 (4KB, converted from f32).
// Per-lane fragment loads follow the ISA 16-bit A/B VGPR layouts.
// ---------------------------------------------------------------------------
#define BM 128
#define BN 64
#define BK 32

__global__ __launch_bounds__(256) void fc1_kernel(
    const _Float16* __restrict__ feat,  // [BATCH][FIN_PAD]  (zero-padded K)
    const float*    __restrict__ w1,    // [NH][FIN]
    const float*    __restrict__ b1,    // [NH]
    _Float16*       __restrict__ hbuf)  // [BATCH][NH_PAD]   (pad cols pre-zeroed)
{
  __shared__ __align__(16) _Float16 lA[BM * BK];
  __shared__ __align__(16) _Float16 lB[BN * BK];

  const int tid  = threadIdx.x;
  const int lane = tid & 31;
  const int wave = tid >> 5;        // 0..7
  const int mw   = wave >> 1;       // 0..3  -> m offset mw*32
  const int nw   = wave & 1;        // 0..1  -> n offset nw*32
  const int m0   = blockIdx.x * BM;
  const int n0   = blockIdx.y * BN;
  const int lHi  = lane >> 4;       // 0/1
  const int lLo  = lane & 15;

  f8 acc[2][2] = {};

  for (int kk = 0; kk < FIN_PAD; kk += BK) {
    __syncthreads();
    // ---- stage A tile: 128 rows x 32 halves = 512 x 16B chunks ----
#pragma unroll
    for (int i = 0; i < 2; ++i) {
      const int idx = tid + 256 * i;
      const int r   = idx >> 2;
      const int ck  = idx & 3;
      const u4 v = *(const u4*)(feat + (size_t)(m0 + r) * FIN_PAD + kk + ck * 8);
      *(u4*)(lA + r * BK + ck * 8) = v;
    }
    // ---- stage B tile: 64 rows x 32 f32 -> f16, bounds-guarded ----
#pragma unroll
    for (int i = 0; i < 2; ++i) {
      const int idx = tid + 256 * i;
      const int n   = idx >> 3;
      const int ck  = idx & 7;
      const int kg  = kk + ck * 4;
      const int ng  = n0 + n;
      f4 v = {0.f, 0.f, 0.f, 0.f};
      if (ng < NH && kg < FIN)
        v = *(const f4*)(w1 + (size_t)ng * FIN + kg);
      h4 hv;
      hv.x = (_Float16)v.x; hv.y = (_Float16)v.y;
      hv.z = (_Float16)v.z; hv.w = (_Float16)v.w;
      *(h4*)(lB + n * BK + ck * 4) = hv;
    }
    // prefetch next A tile strip (global_prefetch_b8)
    if (kk + BK < FIN_PAD)
      __builtin_prefetch(feat + (size_t)(m0 + (tid >> 1)) * FIN_PAD + kk + BK, 0, 3);
    __syncthreads();

    // ---- compute: each wave does a 32x32 patch = 2x2 WMMA tiles ----
#pragma unroll
    for (int mi = 0; mi < 2; ++mi) {
      h16 a;
      const _Float16* ap = lA + (mw * 32 + mi * 16 + lLo) * BK + lHi * 8;
      ((u4*)&a)[0] = *(const u4*)(ap);        // K = kb .. kb+7
      ((u4*)&a)[1] = *(const u4*)(ap + 16);   // K = kb+16 .. kb+23
#pragma unroll
      for (int ni = 0; ni < 2; ++ni) {
        h16 bm;
        const _Float16* bp = lB + (nw * 32 + ni * 16 + lLo) * BK + lHi * 8;
        ((u4*)&bm)[0] = *(const u4*)(bp);     // K = kh*8 .. +7
        ((u4*)&bm)[1] = *(const u4*)(bp + 16); // K = 16+kh*8 .. +7
        acc[mi][ni] = __builtin_amdgcn_wmma_f32_16x16x32_f16(
            false, a, false, bm, (short)0, acc[mi][ni], false, false);
      }
    }
  }

  // ---- epilogue: bias + ReLU, convert to f16 for the fc2 WMMA pass ----
#pragma unroll
  for (int mi = 0; mi < 2; ++mi) {
#pragma unroll
    for (int ni = 0; ni < 2; ++ni) {
      const int n = n0 + nw * 32 + ni * 16 + lLo;
      if (n < NH) {
        const float bias = b1[n];
#pragma unroll
        for (int r = 0; r < 8; ++r) {
          const int m = m0 + mw * 32 + mi * 16 + lHi * 8 + r;
          float v = acc[mi][ni][r] + bias;
          v = fmaxf(v, 0.f);
          hbuf[(size_t)m * NH_PAD + n] = (_Float16)v;
        }
      }
    }
  }
}

// ---------------------------------------------------------------------------
// Kernel 3: fc2 GEMM via WMMA. One wave per 16-row tile; N=14 fits one tile.
// A read straight from global hbuf, B from padded f16 weights (L2-resident).
// ---------------------------------------------------------------------------
__global__ __launch_bounds__(32) void fc2_kernel(
    const _Float16* __restrict__ hbuf,  // [BATCH][NH_PAD]
    const _Float16* __restrict__ w2h,   // [16][NH_PAD] (zero-padded)
    const float*    __restrict__ b2,    // [NCLS]
    float*          __restrict__ out)   // [BATCH][NCLS]
{
  const int m0   = blockIdx.x * 16;
  const int lane = threadIdx.x;
  const int lHi  = lane >> 4;
  const int lLo  = lane & 15;

  f8 acc = {};
  for (int kk = 0; kk < NH_PAD; kk += 32) {
    h16 a, bm;
    const _Float16* ap = hbuf + (size_t)(m0 + lLo) * NH_PAD + kk + lHi * 8;
    ((u4*)&a)[0] = *(const u4*)(ap);
    ((u4*)&a)[1] = *(const u4*)(ap + 16);
    const _Float16* bp = w2h + lLo * NH_PAD + kk + lHi * 8;
    ((u4*)&bm)[0] = *(const u4*)(bp);
    ((u4*)&bm)[1] = *(const u4*)(bp + 16);
    acc = __builtin_amdgcn_wmma_f32_16x16x32_f16(
        false, a, false, bm, (short)0, acc, false, false);
  }
  if (lLo < NCLS) {
    const float bias = b2[lLo];
#pragma unroll
    for (int r = 0; r < 8; ++r)
      out[(size_t)(m0 + lHi * 8 + r) * NCLS + lLo] = acc[r] + bias;
  }
}

// Convert w_fc2 (14 x 1936 f32) into padded 16 x 1952 f16 (zeros elsewhere).
__global__ void cvt_w2_kernel(const float* __restrict__ w2,
                              _Float16* __restrict__ w2h)
{
  const int i = blockIdx.x * 256 + threadIdx.x;
  if (i >= 16 * NH_PAD) return;
  const int n = i / NH_PAD;
  const int k = i - n * NH_PAD;
  const float v = (n < NCLS && k < NH) ? w2[n * NH + k] : 0.f;
  w2h[i] = (_Float16)v;
}

// ---------------------------------------------------------------------------
extern "C" void kernel_launch(void* const* d_in, const int* in_sizes, int n_in,
                              void* d_out, int out_size, void* d_ws, size_t ws_size,
                              hipStream_t stream)
{
  const float* x     = (const float*)d_in[0];
  const float* w_hs  = (const float*)d_in[1];
  const float* b_hs  = (const float*)d_in[2];
  const float* w_ls  = (const float*)d_in[3];
  const float* b_ls  = (const float*)d_in[4];
  const float* w_hi1 = (const float*)d_in[5];
  const float* b_hi1 = (const float*)d_in[6];
  const float* w_hi2 = (const float*)d_in[7];
  const float* b_hi2 = (const float*)d_in[8];
  const float* w_li1 = (const float*)d_in[9];
  const float* b_li1 = (const float*)d_in[10];
  const float* w_li2 = (const float*)d_in[11];
  const float* b_li2 = (const float*)d_in[12];
  const float* w_fc1 = (const float*)d_in[13];
  const float* b_fc1 = (const float*)d_in[14];
  const float* w_fc2 = (const float*)d_in[15];
  const float* b_fc2 = (const float*)d_in[16];
  float* out = (float*)d_out;

  // workspace layout (f16): feat[2048][7136] | hbuf[2048][1952] | w2h[16][1952]
  const size_t FEAT_H = (size_t)BATCH * FIN_PAD;
  const size_t HBUF_H = (size_t)BATCH * NH_PAD;
  _Float16* featb = (_Float16*)d_ws;
  _Float16* hbuf  = featb + FEAT_H;
  _Float16* w2h   = hbuf + HBUF_H;

  // zero pad columns of feat & hbuf (K padding for the WMMA GEMMs)
  hipMemsetAsync(d_ws, 0, (FEAT_H + HBUF_H) * sizeof(_Float16), stream);

  cvt_w2_kernel<<<(16 * NH_PAD + 255) / 256, 256, 0, stream>>>(w_fc2, w2h);

  feat_kernel<<<(BATCH * CC) / 32, 32, 576 * 32 * sizeof(float), stream>>>(
      x, w_hs, b_hs, w_ls, b_ls, w_hi1, b_hi1, w_hi2, b_hi2,
      w_li1, b_li1, w_li2, b_li2, featb);

  dim3 g1(BATCH / BM, (NH + BN - 1) / BN);   // 16 x 31
  fc1_kernel<<<g1, 256, 0, stream>>>(featb, w_fc1, b_fc1, hbuf);

  fc2_kernel<<<BATCH / 16, 32, 0, stream>>>(hbuf, w2h, b_fc2, out);
}
